// MeanPooling_9629316678012
// MI455X (gfx1250) — compile-verified, matrix operand order checked by reading, not run on hardware
//
#include <hip/hip_runtime.h>
#include <hip/hip_bf16.h>

typedef __attribute__((ext_vector_type(16))) __bf16 bf16x16;
typedef __attribute__((ext_vector_type(8)))  __bf16 bf16x8;
typedef __attribute__((ext_vector_type(4)))  __bf16 bf16x4;
typedef __attribute__((ext_vector_type(2)))  __bf16 bf16x2;
typedef __attribute__((ext_vector_type(8)))  float  v8f;
typedef __attribute__((ext_vector_type(4)))  float  f32x4;

#define BB   16
#define EE   128
#define LL   2048
#define HH   1024
#define KT   32          // K per step (bf16 WMMA K)
#define NT   128         // N tile per workgroup
#define LST  40          // LDS row stride in bf16: 32 + 8 pad, keeps 16B alignment
#define NSTEPS (LL / KT) // 64

__global__ __launch_bounds__(256)
void mean_pool_wmma(const float* __restrict__ doc,    // (B, L, H)
                    const float* __restrict__ mask,   // (B, E, L)
                    const float* __restrict__ lens,   // (B, E)
                    float* __restrict__ out)          // (B, E, H)
{
    // double-buffered tiles
    __shared__ __bf16 As[2][EE][LST];   // mask tile (e, k)   -- exact in bf16
    __shared__ __bf16 Bh[2][NT][LST];   // doc tile hi part, transposed (n, k)
    __shared__ __bf16 Bl[2][NT][LST];   // doc tile lo part, transposed (n, k)
    __shared__ float  invLen[EE];

    const int b    = blockIdx.y;
    const int h0   = blockIdx.x * NT;
    const int tid  = threadIdx.x;
    const int lane = tid & 31;
    const int wave = tid >> 5;       // 0..7
    const int wm   = wave & 3;       // M block of 32 rows
    const int wn   = wave >> 2;      // N block of 64 cols
    const int half = lane >> 4;      // 0 or 1
    const int l16  = lane & 15;

    if (tid < EE) invLen[tid] = 1.0f / lens[b * EE + tid];

    const float* maskB = mask + (size_t)b * EE * LL;
    const float* docB  = doc  + (size_t)b * LL * HH + h0;

    // per-thread staging indices (constant over K loop)
    // A: 4 float4 per thread; flat4 = tid + i*256 ; e = flat4/8 ; l = (flat4%8)*4
    // B: 2 row-pairs of float4; pair = tid + i*256 ; l = (pair/32)*2 ; n0 = (pair%32)*4
    int aE[4], aL[4];
#pragma unroll
    for (int i = 0; i < 4; ++i) {
        int flat4 = tid + i * 256;
        aE[i] = flat4 >> 3;
        aL[i] = (flat4 & 7) << 2;
    }
    int bL[2], bN[2];
#pragma unroll
    for (int i = 0; i < 2; ++i) {
        int pair = tid + i * 256;
        bL[i] = (pair >> 5) << 1;
        bN[i] = (pair & 31) << 2;
    }

    f32x4 ra[4];        // staged mask values
    f32x4 rb[2][2];     // staged doc values: [pair][row-in-pair]

    auto loadGlobal = [&](int k0) {
#pragma unroll
        for (int i = 0; i < 4; ++i)
            ra[i] = *(const f32x4*)&maskB[(size_t)aE[i] * LL + k0 + aL[i]];
#pragma unroll
        for (int i = 0; i < 2; ++i) {
            rb[i][0] = *(const f32x4*)&docB[(size_t)(k0 + bL[i]) * HH + bN[i]];
            rb[i][1] = *(const f32x4*)&docB[(size_t)(k0 + bL[i] + 1) * HH + bN[i]];
        }
    };

    auto storeLDS = [&](int p) {
#pragma unroll
        for (int i = 0; i < 4; ++i) {
            bf16x4 v;
#pragma unroll
            for (int j = 0; j < 4; ++j) v[j] = (__bf16)ra[i][j];
            *(bf16x4*)&As[p][aE[i]][aL[i]] = v;     // ds_store_b64
        }
#pragma unroll
        for (int i = 0; i < 2; ++i) {
#pragma unroll
            for (int j = 0; j < 4; ++j) {
                float  x0 = rb[i][0][j], x1 = rb[i][1][j];
                __bf16 h0v = (__bf16)x0,            h1v = (__bf16)x1;
                __bf16 q0  = (__bf16)(x0 - (float)h0v), q1 = (__bf16)(x1 - (float)h1v);
                bf16x2 hv; hv[0] = h0v; hv[1] = h1v;
                bf16x2 lv; lv[0] = q0;  lv[1] = q1;
                *(bf16x2*)&Bh[p][bN[i] + j][bL[i]] = hv;   // ds_store_b32
                *(bf16x2*)&Bl[p][bN[i] + j][bL[i]] = lv;
            }
        }
    };

    v8f acc[2][4];
#pragma unroll
    for (int s = 0; s < 2; ++s)
#pragma unroll
        for (int t = 0; t < 4; ++t)
            acc[s][t] = (v8f)0.0f;

    // ---- prologue: tile 0 ----
    loadGlobal(0);
    storeLDS(0);

    for (int step = 0; step < NSTEPS; ++step) {
        __syncthreads();   // tile `step` visible; prior reads of alt buffer done
        const int p = step & 1;

        // issue global loads for tile step+1 (overlap with WMMA below)
        if (step + 1 < NSTEPS) loadGlobal((step + 1) * KT);
        // L2-latency cover two stages ahead
        if (step + 2 < NSTEPS) {
            __builtin_prefetch(&maskB[(size_t)aE[0] * LL + (step + 2) * KT + aL[0]], 0, 1);
            __builtin_prefetch(&docB[(size_t)((step + 2) * KT + bL[0]) * HH + bN[0]], 0, 1);
        }

        // ---- fragments per documented 16-bit A/B VGPR layout ----
        bf16x16 aF[2], bHF[4], bLF[4];
#pragma unroll
        for (int s = 0; s < 2; ++s) {
            int m = wm * 32 + s * 16 + l16;
            bf16x8 p0 = *(const bf16x8*)&As[p][m][8 * half];       // K = 8h .. 8h+7
            bf16x8 p1 = *(const bf16x8*)&As[p][m][8 * half + 16];  // K = 16+8h ..
            aF[s] = __builtin_shufflevector(p0, p1, 0,1,2,3,4,5,6,7,8,9,10,11,12,13,14,15);
        }
#pragma unroll
        for (int t = 0; t < 4; ++t) {
            int n = wn * 64 + t * 16 + l16;
            bf16x8 h0v = *(const bf16x8*)&Bh[p][n][8 * half];
            bf16x8 h1v = *(const bf16x8*)&Bh[p][n][8 * half + 16];
            bHF[t] = __builtin_shufflevector(h0v, h1v, 0,1,2,3,4,5,6,7,8,9,10,11,12,13,14,15);
            bf16x8 l0v = *(const bf16x8*)&Bl[p][n][8 * half];
            bf16x8 l1v = *(const bf16x8*)&Bl[p][n][8 * half + 16];
            bLF[t] = __builtin_shufflevector(l0v, l1v, 0,1,2,3,4,5,6,7,8,9,10,11,12,13,14,15);
        }

        // ---- 16 WMMAs per wave per K step (lo then hi into same f32 acc) ----
#pragma unroll
        for (int s = 0; s < 2; ++s)
#pragma unroll
            for (int t = 0; t < 4; ++t) {
                acc[s][t] = __builtin_amdgcn_wmma_f32_16x16x32_bf16(
                    false, aF[s], false, bLF[t], (short)0, acc[s][t], false, false);
                acc[s][t] = __builtin_amdgcn_wmma_f32_16x16x32_bf16(
                    false, aF[s], false, bHF[t], (short)0, acc[s][t], false, false);
            }

        // convert + store tile step+1 into the alternate buffer
        if (step + 1 < NSTEPS) storeLDS(p ^ 1);
    }

    // ---- epilogue: scale by 1/len and store (C layout: row = r + 8*half, col = lane%16) ----
    float* outB = out + (size_t)b * EE * HH + h0;
#pragma unroll
    for (int s = 0; s < 2; ++s) {
#pragma unroll
        for (int t = 0; t < 4; ++t) {
            int ncol = wn * 64 + t * 16 + l16;
#pragma unroll
            for (int r = 0; r < 8; ++r) {
                int e = wm * 32 + s * 16 + r + 8 * half;
                outB[(size_t)e * HH + ncol] = acc[s][t][r] * invLen[e];
            }
        }
    }
}

extern "C" void kernel_launch(void* const* d_in, const int* in_sizes, int n_in,
                              void* d_out, int out_size, void* d_ws, size_t ws_size,
                              hipStream_t stream) {
    const float* doc  = (const float*)d_in[0];   // (16, 2048, 1024) fp32
    const float* mask = (const float*)d_in[1];   // (16, 128, 2048) fp32
    const float* lens = (const float*)d_in[2];   // (16, 128) fp32
    float* out = (float*)d_out;                  // (16, 128, 1024) fp32

    dim3 grid(HH / NT, BB);   // (8, 16) workgroups
    mean_pool_wmma<<<grid, 256, 0, stream>>>(doc, mask, lens, out);
}
